// MixMamba_43181601194402
// MI455X (gfx1250) — compile-verified, matrix operand order checked by reading, not run on hardware
//
#include <hip/hip_runtime.h>
#include <hip/hip_bf16.h>
#include <math.h>

// ---------------- problem constants ----------------
#define BDIM   8
#define LDIM   1024
#define HDIM   32
#define DMODEL 384
#define MAMBA  192
#define DINNER 384
#define DSTATE 16
#define DTRANK 12
#define NDIR   4
#define RED    48
#define INNERC 192
#define KC     8
#define MTOT   (BDIM*LDIM)   // 8192 rows for all (b,l) GEMMs

// ---------------- WMMA types / helpers ----------------
typedef __attribute__((ext_vector_type(16))) __bf16 v16bf;
typedef __attribute__((ext_vector_type(8)))  __bf16 v8bf;
typedef __attribute__((ext_vector_type(8)))  float  v8f;

__device__ __forceinline__ float sigmoidf_(float x){ return 1.f/(1.f+__expf(-x)); }
__device__ __forceinline__ float siluf_(float x){ return x*sigmoidf_(x); }
__device__ __forceinline__ float softplusf_(float x){
  return x > 0.f ? x + log1pf(__expf(-x)) : log1pf(__expf(x));
}

// bf16 fragment: two 16B vector loads per lane ({kb..kb+7},{kb+16..kb+23})
__device__ __forceinline__ v16bf load_frag_bf(const __bf16* p){
  v8bf lo = *(const v8bf*)p;
  v8bf hi = *(const v8bf*)(p + 16);
  v16bf f;
#pragma unroll
  for (int j=0;j<8;++j){ f[j] = lo[j]; f[8+j] = hi[j]; }
  return f;
}

// One wave computes a 16 x (16*NSUB) strip of C = A(MxK) * W(NxK)^T, bf16 in,
// f32 accum. Software-pipelined: fragments for step k+32 are loaded before the
// WMMAs consuming step k. Branch-free N edge: clamp row, mask at store.
template<int NSUB>
__device__ __forceinline__ void wmma_strip(const __bf16* A, int lda,
                                           const __bf16* W, int ldw,
                                           int K, int m0, int n0, int n_valid,
                                           v8f* out){
  const int lane = threadIdx.x & 31;
  const int r    = lane & 15;
  const int kb   = (lane >> 4) << 3;
  const __bf16* arow = A + (size_t)(m0 + r)*lda + kb;
  const __bf16* brow[NSUB];
#pragma unroll
  for (int s=0;s<NSUB;++s){
    int n = n0 + s*16 + r;
    if (n > n_valid-1) n = n_valid-1;
    brow[s] = W + (size_t)n*ldw + kb;
  }
  v8f c0 = {}, c1 = {}, c2 = {}, c3 = {};
  v16bf a  = load_frag_bf(arow);
  v16bf b0 = load_frag_bf(brow[0]);
  v16bf b1 = {}, b2 = {}, b3 = {};
  if (NSUB > 1) b1 = load_frag_bf(brow[1]);
  if (NSUB > 2) b2 = load_frag_bf(brow[2]);
  if (NSUB > 3) b3 = load_frag_bf(brow[3]);
  for (int k0 = 32; k0 < K; k0 += 32){
    // issue next-step loads before consuming current fragments
    v16bf an  = load_frag_bf(arow + k0);
    v16bf bn0 = load_frag_bf(brow[0] + k0);
    v16bf bn1 = {}, bn2 = {}, bn3 = {};
    if (NSUB > 1) bn1 = load_frag_bf(brow[1] + k0);
    if (NSUB > 2) bn2 = load_frag_bf(brow[2] + k0);
    if (NSUB > 3) bn3 = load_frag_bf(brow[3] + k0);
    c0 = __builtin_amdgcn_wmma_f32_16x16x32_bf16(false, a, false, b0, (short)0, c0, false, false);
    if (NSUB > 1)
      c1 = __builtin_amdgcn_wmma_f32_16x16x32_bf16(false, a, false, b1, (short)0, c1, false, false);
    if (NSUB > 2)
      c2 = __builtin_amdgcn_wmma_f32_16x16x32_bf16(false, a, false, b2, (short)0, c2, false, false);
    if (NSUB > 3)
      c3 = __builtin_amdgcn_wmma_f32_16x16x32_bf16(false, a, false, b3, (short)0, c3, false, false);
    a = an; b0 = bn0;
    if (NSUB > 1) b1 = bn1;
    if (NSUB > 2) b2 = bn2;
    if (NSUB > 3) b3 = bn3;
  }
  c0 = __builtin_amdgcn_wmma_f32_16x16x32_bf16(false, a, false, b0, (short)0, c0, false, false);
  if (NSUB > 1)
    c1 = __builtin_amdgcn_wmma_f32_16x16x32_bf16(false, a, false, b1, (short)0, c1, false, false);
  if (NSUB > 2)
    c2 = __builtin_amdgcn_wmma_f32_16x16x32_bf16(false, a, false, b2, (short)0, c2, false, false);
  if (NSUB > 3)
    c3 = __builtin_amdgcn_wmma_f32_16x16x32_bf16(false, a, false, b3, (short)0, c3, false, false);
  out[0] = c0;
  if (NSUB > 1) out[1] = c1;
  if (NSUB > 2) out[2] = c2;
  if (NSUB > 3) out[3] = c3;
}

// direction index map: sequence position p -> canonical l
__device__ __forceinline__ int lmap(int dir, int p){
  switch (dir){
    case 0:  return p;
    case 1:  return LDIM-1-p;
    case 2:  return (p & 31)*32 + (p >> 5);            // 32x32 transpose
    default: { int q = LDIM-1-p; return (q & 31)*32 + (q >> 5); }
  }
}

// ---------------- staging: f32 -> bf16 ----------------
__global__ void k_f2bf(const float* __restrict__ s, __bf16* __restrict__ d, int n){
  int i = blockIdx.x*256 + threadIdx.x;
  if (i < n) d[i] = (__bf16)s[i];
}

// ---------------- GEMM kernels (WMMA bf16, f32 accum) ----------------
__global__ void k_gemm_inproj(const __bf16* __restrict__ xbf,
                              const __bf16* __restrict__ w,
                              float* __restrict__ xz){
  int job = blockIdx.x*8 + (threadIdx.x >> 5);       // 512 mt * 12 groups
  int mt = job / 12, g = job % 12;
  if (mt >= MTOT/16) return;
  v8f c[4];
  wmma_strip<4>(xbf, DMODEL, w, MAMBA, MAMBA, mt*16, g*64, 2*DINNER, c);
  int lane = threadIdx.x & 31;
  int mb = mt*16 + ((lane>>4)<<3);
#pragma unroll
  for (int s=0;s<4;++s){
    int n = g*64 + s*16 + (lane&15);
#pragma unroll
    for (int r=0;r<8;++r) xz[(size_t)(mb+r)*(2*DINNER) + n] = c[s][r];
  }
}

__global__ void k_gemm_xproj(const __bf16* __restrict__ ubf,
                             const __bf16* __restrict__ w,
                             float* __restrict__ dbl){
  int dir = blockIdx.y;
  int mt  = blockIdx.x*8 + (threadIdx.x >> 5);       // 512 strips, NSUB=3 covers N=48
  if (mt >= MTOT/16) return;
  const __bf16* A = ubf + (size_t)dir*MTOT*DINNER;
  const __bf16* W = w   + (size_t)dir*(DTRANK+2*DSTATE)*DINNER;
  v8f c[3];
  wmma_strip<3>(A, DINNER, W, DINNER, DINNER, mt*16, 0, DTRANK+2*DSTATE, c);
  int lane = threadIdx.x & 31;
  int mb = mt*16 + ((lane>>4)<<3);
#pragma unroll
  for (int s=0;s<3;++s){
    int n = s*16 + (lane&15);
    if (n < DTRANK+2*DSTATE){
#pragma unroll
      for (int r=0;r<8;++r) dbl[((size_t)dir*MTOT + mb + r)*48 + n] = c[s][r];
    }
  }
}

__global__ void k_gemm_outproj(const __bf16* __restrict__ accbf,
                               const __bf16* __restrict__ w,
                               float* __restrict__ out){
  int job = blockIdx.x*8 + (threadIdx.x >> 5);       // 512 mt * 3 groups
  int mt = job / 3, g = job % 3;
  if (mt >= MTOT/16) return;
  v8f c[4];
  wmma_strip<4>(accbf, DINNER, w, DINNER, DINNER, mt*16, g*64, MAMBA, c);
  int lane = threadIdx.x & 31;
  int mb = mt*16 + ((lane>>4)<<3);
#pragma unroll
  for (int s=0;s<4;++s){
    int n = g*64 + s*16 + (lane&15);
#pragma unroll
    for (int r=0;r<8;++r) out[(size_t)(mb+r)*DMODEL + n] = c[s][r];
  }
}

__global__ void k_gemm_pw1(const __bf16* __restrict__ xbf,
                           const __bf16* __restrict__ w,
                           const float* __restrict__ bias,
                           float* __restrict__ hpre){
  int job = blockIdx.x*8 + (threadIdx.x >> 5);       // 512 mt * 6 groups
  int mt = job / 6, g = job % 6;
  if (mt >= MTOT/16) return;
  v8f c[4];
  wmma_strip<4>(xbf + MAMBA, DMODEL, w, INNERC, INNERC, mt*16, g*64, 2*INNERC, c);
  int lane = threadIdx.x & 31;
  int mb = mt*16 + ((lane>>4)<<3);
#pragma unroll
  for (int s=0;s<4;++s){
    int n = g*64 + s*16 + (lane&15);
    float bv = bias[n];
#pragma unroll
    for (int r=0;r<8;++r) hpre[(size_t)(mb+r)*(2*INNERC) + n] = c[s][r] + bv;
  }
}

__global__ void k_gemm_pw2(const __bf16* __restrict__ hbf,
                           const __bf16* __restrict__ w,
                           const float* __restrict__ bias,
                           float* __restrict__ out){
  int job = blockIdx.x*8 + (threadIdx.x >> 5);       // 512 mt * 3 groups
  int mt = job / 3, g = job % 3;
  if (mt >= MTOT/16) return;
  v8f c[4];
  wmma_strip<4>(hbf, INNERC, w, INNERC, INNERC, mt*16, g*64, INNERC, c);
  int lane = threadIdx.x & 31;
  int mb = mt*16 + ((lane>>4)<<3);
#pragma unroll
  for (int s=0;s<4;++s){
    int n = g*64 + s*16 + (lane&15);
    float bv = bias[n];
#pragma unroll
    for (int r=0;r<8;++r) out[(size_t)(mb+r)*DMODEL + MAMBA + n] = c[s][r] + bv;
  }
}

// ---------------- mamba branch: conv + scan + biattn ----------------
// u[dir,b,p,d] = silu(causal conv4 over xp); writes f32 (scan) + bf16 (xproj GEMM)
__global__ void k_conv_u(const float* __restrict__ xz,
                         const float* __restrict__ cw,
                         const float* __restrict__ cb,
                         float* __restrict__ u, __bf16* __restrict__ ubf){
  size_t idx = (size_t)blockIdx.x*blockDim.x + threadIdx.x;
  if (idx >= (size_t)NDIR*BDIM*LDIM*DINNER) return;
  int d   = idx % DINNER;
  int p   = (idx / DINNER) % LDIM;
  int b   = (idx / ((size_t)DINNER*LDIM)) % BDIM;
  int dir =  idx / ((size_t)DINNER*LDIM*BDIM);
  float acc = cb[dir*DINNER + d];
#pragma unroll
  for (int t=0;t<4;++t){
    int pp = p - 3 + t;
    if (pp >= 0){
      float xv = xz[((size_t)b*LDIM + lmap(dir, pp))*(2*DINNER) + d];
      acc += cw[(dir*DINNER + d)*4 + t] * xv;
    }
  }
  float v = siluf_(acc);
  u[idx] = v;
  ubf[idx] = (__bf16)v;
}

// one block = one (dir,b) group, 384 threads = one d channel each.
// dbl row staged in LDS once per step; dt-proj + softplus + scan fused.
__global__ void k_scan(const float* __restrict__ xz,
                       const float* __restrict__ u,
                       const float* __restrict__ dbl,
                       const float* __restrict__ A_log,
                       const float* __restrict__ dtw_g,
                       const float* __restrict__ dtb_g,
                       const float* __restrict__ Dp,
                       float* __restrict__ ydir){
  __shared__ float row_s[48];
  int grp = blockIdx.x;                 // dir*8 + b
  int dir = grp >> 3, b = grp & 7;
  int d   = threadIdx.x;                // 0..383

  float A[DSTATE], h[DSTATE];
#pragma unroll
  for (int n=0;n<DSTATE;++n){
    A[n] = -__expf(A_log[((size_t)dir*DINNER + d)*DSTATE + n]);
    h[n] = 0.f;
  }
  float dtw[DTRANK];
#pragma unroll
  for (int r=0;r<DTRANK;++r) dtw[r] = dtw_g[((size_t)dir*DINNER + d)*DTRANK + r];
  const float dtb = dtb_g[dir*DINNER + d];
  const float Dv  = Dp[dir*DINNER + d];

  const float* ub = u   + (((size_t)dir*BDIM + b)*LDIM)*DINNER + d;
  const float* db = dbl + (((size_t)dir*BDIM + b)*LDIM)*48;

  for (int p=0;p<LDIM;++p){
    if (threadIdx.x < 44) row_s[threadIdx.x] = db[(size_t)p*48 + threadIdx.x];
    if (threadIdx.x == 0 && p+8 < LDIM)
      __builtin_prefetch((const void*)(db + (size_t)(p+8)*48), 0, 0);
    __syncthreads();
    float dt = dtb;
#pragma unroll
    for (int r=0;r<DTRANK;++r) dt += dtw[r]*row_s[r];
    float delta = softplusf_(dt);
    float uv = ub[(size_t)p*DINNER];
    float du = delta*uv;
    float y = 0.f;
#pragma unroll
    for (int n=0;n<DSTATE;++n){
      float e = __expf(delta*A[n]);
      h[n] = e*h[n] + du*row_s[DTRANK + n];
      y   += h[n]*row_s[DTRANK + DSTATE + n];
    }
    y += uv*Dv;
    int l = lmap(dir, p);
    float zv = xz[((size_t)b*LDIM + l)*(2*DINNER) + DINNER + d];
    ydir[(((size_t)dir*BDIM + b)*LDIM + l)*DINNER + d] = y * siluf_(zv);
    __syncthreads();
  }
}

// per (dir,b,l) LayerNorm statistics
__global__ void k_lnstats(const float* __restrict__ ydir,
                          float* __restrict__ mu, float* __restrict__ rstd){
  int idx = blockIdx.x*blockDim.x + threadIdx.x;    // 4*8192 rows
  if (idx >= NDIR*MTOT) return;
  const float* row = ydir + (size_t)idx*DINNER;
  float s=0.f, s2=0.f;
  for (int dd=0; dd<DINNER; ++dd){ float v=row[dd]; s+=v; s2+=v*v; }
  float m = s*(1.f/DINNER);
  float var = s2*(1.f/DINNER) - m*m;
  mu[idx] = m; rstd[idx] = rsqrtf(var + 1e-5f);
}

// gmean[dir,b,d] = mean_l (y - mu)*rstd     (affine applied in gate kernel)
__global__ void k_gmean(const float* __restrict__ ydir,
                        const float* __restrict__ mu, const float* __restrict__ rstd,
                        float* __restrict__ gmean){
  int idx = blockIdx.x*blockDim.x + threadIdx.x;    // 4*8*384
  if (idx >= NDIR*BDIM*DINNER) return;
  int d   = idx % DINNER;
  int b   = (idx / DINNER) % BDIM;
  int dir =  idx / (DINNER*BDIM);
  size_t rb = ((size_t)dir*BDIM + b)*LDIM;
  float s = 0.f;
  for (int l=0; l<LDIM; ++l)
    s += (ydir[(rb+l)*DINNER + d] - mu[rb+l]) * rstd[rb+l];
  gmean[idx] = s*(1.f/LDIM);
}

// per (dir,b): gelu(GR) -> sigmoid(CS) channel gate
__global__ void k_gate(const float* __restrict__ gmean,
                       const float* __restrict__ lng, const float* __restrict__ lnb,
                       const float* __restrict__ grw, const float* __restrict__ grb,
                       const float* __restrict__ csw, const float* __restrict__ csb,
                       float* __restrict__ cgate){
  __shared__ float gm[DINNER];
  __shared__ float g1[RED];
  int dir = blockIdx.x / BDIM, b = blockIdx.x % BDIM;
  int tid = threadIdx.x;
  if (tid < DINNER)
    gm[tid] = gmean[((size_t)dir*BDIM + b)*DINNER + tid]*lng[tid] + lnb[tid];
  __syncthreads();
  if (tid < RED){
    float v = grb[tid];
    for (int dd=0; dd<DINNER; ++dd) v += gm[dd]*grw[(size_t)tid*DINNER + dd];
    g1[tid] = 0.5f*v*(1.f + erff(v*0.70710678118f));      // exact gelu
  }
  __syncthreads();
  if (tid < DINNER){
    float s = csb[tid];
#pragma unroll
    for (int r=0;r<RED;++r) s += g1[r]*csw[(size_t)tid*RED + r];
    cgate[((size_t)dir*BDIM + b)*DINNER + tid] = sigmoidf_(s);
  }
}

// acc[b,l,d] = sum_dir ydir * cgate   -> bf16 (feeds out_proj WMMA)
__global__ void k_accsum(const float* __restrict__ ydir,
                         const float* __restrict__ cgate,
                         __bf16* __restrict__ accbf){
  size_t idx = (size_t)blockIdx.x*blockDim.x + threadIdx.x;  // 8192*384
  if (idx >= (size_t)MTOT*DINNER) return;
  int d = idx % DINNER;
  size_t m = idx / DINNER;
  int b = (int)(m / LDIM);
  float s = 0.f;
#pragma unroll
  for (int dir=0;dir<NDIR;++dir)
    s += ydir[((size_t)dir*MTOT + m)*DINNER + d] *
         cgate[((size_t)dir*BDIM + b)*DINNER + d];
  accbf[idx] = (__bf16)s;
}

// ---------------- local branch ----------------
__global__ void k_glu(const float* __restrict__ hpre, float* __restrict__ hglu){
  size_t idx = (size_t)blockIdx.x*blockDim.x + threadIdx.x;  // 8192*192
  if (idx >= (size_t)MTOT*INNERC) return;
  int c = idx % INNERC;
  size_t m = idx / INNERC;
  float a = hpre[m*(2*INNERC) + c];
  float g = hpre[m*(2*INNERC) + INNERC + c];
  hglu[idx] = a*sigmoidf_(g);
}

// depthwise conv K=8 (pad 4/3) + per-channel BN partial stats
__global__ void k_dwconv(const float* __restrict__ hglu,
                         const float* __restrict__ dww, const float* __restrict__ dwb,
                         float* __restrict__ hconv, float* __restrict__ stats){
  __shared__ float s1[256], s2[256];
  int c = blockIdx.x % INNERC, b = blockIdx.x / INNERC;
  int l = blockIdx.y*256 + threadIdx.x;
  float o = dwb[c];
#pragma unroll
  for (int t=0;t<KC;++t){
    int li = l - 4 + t;
    if (li >= 0 && li < LDIM)
      o += dww[c*KC + t]*hglu[((size_t)b*LDIM + li)*INNERC + c];
  }
  hconv[((size_t)b*LDIM + l)*INNERC + c] = o;
  s1[threadIdx.x] = o; s2[threadIdx.x] = o*o;
  __syncthreads();
  for (int off=128; off>0; off>>=1){
    if (threadIdx.x < off){ s1[threadIdx.x]+=s1[threadIdx.x+off];
                            s2[threadIdx.x]+=s2[threadIdx.x+off]; }
    __syncthreads();
  }
  if (threadIdx.x == 0){
    atomicAdd(&stats[c], s1[0]);
    atomicAdd(&stats[INNERC + c], s2[0]);
  }
}

__global__ void k_bnfinal(const float* __restrict__ bng, const float* __restrict__ bnb,
                          float* __restrict__ stats){
  int c = threadIdx.x;
  if (c >= INNERC) return;
  const float inv = 1.f/(float)(BDIM*LDIM);
  float m = stats[c]*inv;
  float var = stats[INNERC + c]*inv - m*m;
  float rs = rsqrtf(var + 1e-5f);
  float sc = bng[c]*rs;
  stats[2*INNERC + c] = sc;
  stats[3*INNERC + c] = bnb[c] - m*sc;
}

// BN + silu -> bf16 (feeds pw2 WMMA)
__global__ void k_bnapply(const float* __restrict__ hconv,
                          const float* __restrict__ stats,
                          __bf16* __restrict__ hbf){
  size_t idx = (size_t)blockIdx.x*blockDim.x + threadIdx.x;
  if (idx >= (size_t)MTOT*INNERC) return;
  int c = idx % INNERC;
  float v = hconv[idx]*stats[2*INNERC + c] + stats[3*INNERC + c];
  hbf[idx] = (__bf16)(v*sigmoidf_(v));
}

// ---------------- launcher ----------------
extern "C" void kernel_launch(void* const* d_in, const int* in_sizes, int n_in,
                              void* d_out, int out_size, void* d_ws, size_t ws_size,
                              hipStream_t stream){
  const float* x    = (const float*)d_in[0];
  const float* ipw  = (const float*)d_in[1];
  const float* Alog = (const float*)d_in[2];
  const float* cw   = (const float*)d_in[3];
  const float* cb   = (const float*)d_in[4];
  const float* xpw  = (const float*)d_in[5];
  const float* dtw  = (const float*)d_in[6];
  const float* dtb  = (const float*)d_in[7];
  const float* Dp   = (const float*)d_in[8];
  const float* opw  = (const float*)d_in[9];
  const float* lng  = (const float*)d_in[10];
  const float* lnb  = (const float*)d_in[11];
  const float* grw  = (const float*)d_in[12];
  const float* grb  = (const float*)d_in[13];
  const float* csw  = (const float*)d_in[14];
  const float* csb  = (const float*)d_in[15];
  const float* p1w  = (const float*)d_in[16];
  const float* p1b  = (const float*)d_in[17];
  const float* dww  = (const float*)d_in[18];
  const float* dwb  = (const float*)d_in[19];
  const float* bng  = (const float*)d_in[20];
  const float* bnb  = (const float*)d_in[21];
  const float* p2w  = (const float*)d_in[22];
  const float* p2b  = (const float*)d_in[23];
  float* out = (float*)d_out;

  float* ws = (float*)d_ws;
  float* xz    = ws;                        //  8192*768
  float* u     = ws + 6291456;              //  4*8192*384
  float* dbl   = ws + 18874368;             //  4*8192*48
  float* ydir  = ws + 20447232;             //  4*8192*384 (reused by local branch)
  float* mu    = ws + 33030144;             //  32768
  float* rstd  = ws + 33062912;             //  32768
  float* gmean = ws + 33095680;             //  12288
  float* cgate = ws + 33107968;             //  12288
  float* stats = ws + 33120256;             //  1024
  float* hpre  = ydir;                      // ydir dead after accsum
  float* hglu  = ydir + 3145728;
  float* hconv = ydir + 4718592;

  // bf16 staging region (16B aligned)
  __bf16* wsbf   = (__bf16*)(ws + 33121280);
  __bf16* ipw_bf = wsbf;                    // 768*192   = 147456
  __bf16* xpw_bf = wsbf + 147456;           // 4*44*384  =  67584
  __bf16* opw_bf = wsbf + 215040;           // 192*384   =  73728
  __bf16* p1w_bf = wsbf + 288768;           // 384*192   =  73728
  __bf16* p2w_bf = wsbf + 362496;           // 192*192   =  36864
  __bf16* x_bf   = wsbf + 399360;           // 8192*384
  __bf16* u_bf   = wsbf + 3545088;          // 4*8192*384
  __bf16* acc_bf = u_bf;                    // u_bf dead after xproj GEMM
  __bf16* hbn_bf = u_bf + 3145728;

  // weight + activation staging
  k_f2bf<<<576, 256, 0, stream>>>(ipw, ipw_bf, 147456);
  k_f2bf<<<264, 256, 0, stream>>>(xpw, xpw_bf, 67584);
  k_f2bf<<<288, 256, 0, stream>>>(opw, opw_bf, 73728);
  k_f2bf<<<288, 256, 0, stream>>>(p1w, p1w_bf, 73728);
  k_f2bf<<<144, 256, 0, stream>>>(p2w, p2w_bf, 36864);
  k_f2bf<<<12288, 256, 0, stream>>>(x, x_bf, 3145728);

  // mamba branch
  k_gemm_inproj<<<768, 256, 0, stream>>>(x_bf, ipw_bf, xz);
  k_conv_u<<<49152, 256, 0, stream>>>(xz, cw, cb, u, u_bf);
  k_gemm_xproj<<<dim3(64, NDIR), 256, 0, stream>>>(u_bf, xpw_bf, dbl);
  k_scan<<<NDIR*BDIM, 384, 0, stream>>>(xz, u, dbl, Alog, dtw, dtb, Dp, ydir);
  k_lnstats<<<128, 256, 0, stream>>>(ydir, mu, rstd);
  k_gmean<<<48, 256, 0, stream>>>(ydir, mu, rstd, gmean);
  k_gate<<<NDIR*BDIM, 384, 0, stream>>>(gmean, lng, lnb, grw, grb, csw, csb, cgate);
  k_accsum<<<12288, 256, 0, stream>>>(ydir, cgate, acc_bf);
  k_gemm_outproj<<<192, 256, 0, stream>>>(acc_bf, opw_bf, out);

  // local branch
  k_gemm_pw1<<<384, 256, 0, stream>>>(x_bf, p1w_bf, p1b, hpre);
  k_glu<<<6144, 256, 0, stream>>>(hpre, hglu);
  hipMemsetAsync(stats, 0, 2*INNERC*sizeof(float), stream);
  k_dwconv<<<dim3(BDIM*INNERC, LDIM/256), 256, 0, stream>>>(hglu, dww, dwb, hconv, stats);
  k_bnfinal<<<1, 256, 0, stream>>>(bng, bnb, stats);
  k_bnapply<<<6144, 256, 0, stream>>>(hconv, stats, hbn_bf);
  k_gemm_pw2<<<192, 256, 0, stream>>>(hbn_bf, p2w_bf, p2b, out);
}
